// Transformer_85452669321844
// MI455X (gfx1250) — matrix operand order repaired and checked by reading, not compiled, not run on hardware
//
#include <hip/hip_runtime.h>

// ---------------------------------------------------------------------------
// Transformer encoder-decoder forward for MI455X (gfx1250, wave32, WMMA).
// GEMMs: bf16 inputs, fp32 accumulation via v_wmma_f32_16x16x32_bf16,
// global->LDS staging via CDNA5 async-to-LDS (ASYNCcnt) with double buffering.
// ---------------------------------------------------------------------------

#define DMODEL 768
#define NHEAD  12
#define DHEAD  64
#define FFNH   3072
#define NLAYER 6
#define SEQ    1024

typedef __bf16 bf16_t;
typedef bf16_t v16bf __attribute__((ext_vector_type(16)));
typedef float  v8f   __attribute__((ext_vector_type(8)));

__device__ __forceinline__ unsigned short f32_to_bf16(float f) {
  unsigned int u = __float_as_uint(f);
  u += 0x7FFFu + ((u >> 16) & 1u);          // round-to-nearest-even
  return (unsigned short)(u >> 16);
}

// CDNA5 async global->LDS copy, 16 bytes per lane. LDS operand is the
// wave-relative LDS byte offset == low 32 bits of the generic LDS pointer
// (ISA 10.2).
__device__ __forceinline__ void async_load_b128(const void* gptr, void* lptr) {
  unsigned lds_off = (unsigned)(unsigned long long)lptr;
  asm volatile("global_load_async_to_lds_b128 %0, %1, off"
               :: "v"(lds_off), "v"(gptr)
               : "memory");
}
__device__ __forceinline__ void wait_async0() {
  asm volatile("s_wait_asynccnt 0x0" ::: "memory");
}

// ---------------------------------------------------------------------------
// WMMA GEMM:  C[M,N] = A[M,K] * B[N,K]^T (+bias) (optional relu)
// A, B are bf16 (raw u16) row-major with leading dims lda/ldb.
// Optional fp32 output C32 and/or bf16 output Cbf (same ldc).
// Batched over gridDim.z with element strides sA/sB/sC.
// Block tile: 128x64x32, 8 waves, each wave 32x32 via 2x2 WMMA fragments.
// Double-buffered LDS; tile i+1 transfers overlap tile i WMMAs.
// ---------------------------------------------------------------------------
#define BM 128
#define BN 64
#define BK 32
#define LDSA 40   // padded row stride (halves), multiple of 8 for 16B ops
#define LDSB 40

union Frag { uint4 q[2]; v16bf v; };

__global__ __launch_bounds__(256)
void k_gemm_bf16nt(const unsigned short* __restrict__ A,
                   const unsigned short* __restrict__ B,
                   const float* __restrict__ bias,
                   float* __restrict__ C32,
                   unsigned short* __restrict__ Cbf,
                   int M, int N, int K,
                   int lda, int ldb, int ldc,
                   long sA, long sB, long sC, int relu)
{
  (void)M; (void)N;
  __shared__ __align__(16) unsigned short As[2][BM * LDSA];
  __shared__ __align__(16) unsigned short Bs[2][BN * LDSB];

  const int bz = blockIdx.z;
  A += (long)bz * sA;
  B += (long)bz * sB;
  const long cOff = (long)bz * sC;

  const int bm   = blockIdx.y * BM;
  const int bn   = blockIdx.x * BN;
  const int tid  = threadIdx.x;
  const int lane = tid & 31;
  const int wave = tid >> 5;
  const int wm   = (wave & 3) * 32;   // wave M offset in tile
  const int wn   = (wave >> 2) * 32;  // wave N offset in tile
  const int frow  = lane & 15;        // fragment row (A) / col (B)
  const int fhalf = lane >> 4;        // lane half selects K sub-range

  v8f acc[2][2] = {};

  const int lr = tid >> 2;            // 0..63  (staging row)
  const int lc = (tid & 3) * 8;       // 0,8,16,24 (staging col, halves)

  // issue async transfers of one K-slab into LDS buffer bi
  auto stage = [&](int k0, int bi) {
    const unsigned short* ga = A + (long)(bm + lr) * lda + (k0 + lc);
    async_load_b128(ga,                    &As[bi][lr * LDSA + lc]);
    async_load_b128(ga + (long)64 * lda,   &As[bi][(lr + 64) * LDSA + lc]);
    const unsigned short* gb = B + (long)(bn + lr) * ldb + (k0 + lc);
    async_load_b128(gb,                    &Bs[bi][lr * LDSB + lc]);
  };

  const int nk = K / BK;
  stage(0, 0);

  for (int i = 0; i < nk; ++i) {
    wait_async0();        // my async writes for buffer i&1 have landed
    __syncthreads();      // everyone's landed; prior reads of other buf done
    if (i + 1 < nk) stage((i + 1) * BK, (i + 1) & 1);   // overlap next slab

    const int bi = i & 1;
    // ---- fragment loads per ISA 7.12.2 layouts ----
    Frag a0, a1, b0, b1;
    {
      const unsigned short* p0 = &As[bi][(wm + frow) * LDSA];
      a0.q[0] = *(const uint4*)(p0 + fhalf * 8);
      a0.q[1] = *(const uint4*)(p0 + 16 + fhalf * 8);
      const unsigned short* p1 = &As[bi][(wm + 16 + frow) * LDSA];
      a1.q[0] = *(const uint4*)(p1 + fhalf * 8);
      a1.q[1] = *(const uint4*)(p1 + 16 + fhalf * 8);
      const unsigned short* q0 = &Bs[bi][(wn + frow) * LDSB + fhalf * 8];
      b0.q[0] = *(const uint4*)(q0);
      b0.q[1] = *(const uint4*)(q0 + 16);
      const unsigned short* q1 = &Bs[bi][(wn + 16 + frow) * LDSB + fhalf * 8];
      b1.q[0] = *(const uint4*)(q1);
      b1.q[1] = *(const uint4*)(q1 + 16);
    }

    acc[0][0] = __builtin_amdgcn_wmma_f32_16x16x32_bf16(false, a0.v, false, b0.v, (short)0, acc[0][0], false, false);
    acc[0][1] = __builtin_amdgcn_wmma_f32_16x16x32_bf16(false, a0.v, false, b1.v, (short)0, acc[0][1], false, false);
    acc[1][0] = __builtin_amdgcn_wmma_f32_16x16x32_bf16(false, a1.v, false, b0.v, (short)0, acc[1][0], false, false);
    acc[1][1] = __builtin_amdgcn_wmma_f32_16x16x32_bf16(false, a1.v, false, b1.v, (short)0, acc[1][1], false, false);
  }

  // ---- epilogue: bias + relu, fp32 and/or bf16 stores ----
  // C layout (32-bit 16x16): lane = (m>>3)*16 + n ; VGPR v -> M = (lane>>4)*8+v
  #pragma unroll
  for (int i = 0; i < 2; ++i) {
    #pragma unroll
    for (int j = 0; j < 2; ++j) {
      const int n  = bn + wn + j * 16 + frow;
      const int mb = bm + wm + i * 16 + fhalf * 8;
      const float bv = bias ? bias[n] : 0.0f;
      #pragma unroll
      for (int v = 0; v < 8; ++v) {
        float val = acc[i][j][v] + bv;
        if (relu) val = fmaxf(val, 0.0f);
        const long idx = cOff + (long)(mb + v) * ldc + n;
        if (C32) C32[idx] = val;
        if (Cbf) Cbf[idx] = f32_to_bf16(val);
      }
    }
  }
}

// ---------------------------------------------------------------------------
// Elementwise / reduction kernels
// ---------------------------------------------------------------------------
__global__ __launch_bounds__(256)
void k_f32_to_bf16(const float* __restrict__ in, unsigned short* __restrict__ out, int n) {
  int i = blockIdx.x * 256 + threadIdx.x;
  if (i < n) out[i] = f32_to_bf16(in[i]);
}

__global__ __launch_bounds__(256)
void k_add_pos(const float* __restrict__ x, const float* __restrict__ pe,
               float* __restrict__ oF, unsigned short* __restrict__ oB, int D) {
  const long base = (long)blockIdx.x * D;
  for (int c = threadIdx.x; c < D; c += 256) {
    float v = x[base + c] + pe[base + c];
    oF[base + c] = v;
    oB[base + c] = f32_to_bf16(v);
  }
}

// V [SEQ][DMODEL] bf16 -> Vt [DMODEL][SEQ] bf16  (per-head column-major for AV)
__global__ __launch_bounds__(256)
void k_transpose_tv(const unsigned short* __restrict__ V, unsigned short* __restrict__ Vt) {
  const int idx = blockIdx.x * 256 + threadIdx.x;  // 0 .. SEQ*DMODEL-1
  const int t = idx / DMODEL;
  const int c = idx - t * DMODEL;
  Vt[(long)c * SEQ + t] = V[idx];
}

// scores [H][S][T] fp32 -> P bf16, scale + optional causal mask, softmax over T
__global__ __launch_bounds__(256)
void k_attn_softmax(const float* __restrict__ S, unsigned short* __restrict__ P,
                    int T, int causal, float scale) {
  __shared__ float red[256];
  const int row = blockIdx.x;
  const long base = ((long)blockIdx.y * gridDim.x + row) * T;
  const int tid = threadIdx.x;
  float mx = -3.4e38f;
  for (int c = tid; c < T; c += 256) {
    float v = S[base + c] * scale;
    if (causal && c > row) v -= 10000.0f;
    mx = fmaxf(mx, v);
  }
  red[tid] = mx; __syncthreads();
  for (int o = 128; o > 0; o >>= 1) { if (tid < o) red[tid] = fmaxf(red[tid], red[tid + o]); __syncthreads(); }
  mx = red[0]; __syncthreads();
  float s = 0.f;
  for (int c = tid; c < T; c += 256) {
    float v = S[base + c] * scale;
    if (causal && c > row) v -= 10000.0f;
    s += __expf(v - mx);
  }
  red[tid] = s; __syncthreads();
  for (int o = 128; o > 0; o >>= 1) { if (tid < o) red[tid] += red[tid + o]; __syncthreads(); }
  const float inv = 1.0f / red[0];
  for (int c = tid; c < T; c += 256) {
    float v = S[base + c] * scale;
    if (causal && c > row) v -= 10000.0f;
    P[base + c] = f32_to_bf16(__expf(v - mx) * inv);
  }
}

// x = LN(x + fx) with unbiased std (ddof=1), writes fp32 + bf16 copies
__global__ __launch_bounds__(256)
void k_ln_res(float* __restrict__ x, unsigned short* __restrict__ xB,
              const float* __restrict__ fx,
              const float* __restrict__ g, const float* __restrict__ b, int D) {
  __shared__ float red[256];
  const long base = (long)blockIdx.x * D;
  const int tid = threadIdx.x;
  float s = 0.f;
  for (int c = tid; c < D; c += 256) {
    float v = x[base + c] + fx[base + c];
    x[base + c] = v;
    s += v;
  }
  red[tid] = s; __syncthreads();
  for (int o = 128; o > 0; o >>= 1) { if (tid < o) red[tid] += red[tid + o]; __syncthreads(); }
  const float mean = red[0] / (float)D;
  __syncthreads();
  float vs = 0.f;
  for (int c = tid; c < D; c += 256) { float d = x[base + c] - mean; vs += d * d; }
  red[tid] = vs; __syncthreads();
  for (int o = 128; o > 0; o >>= 1) { if (tid < o) red[tid] += red[tid + o]; __syncthreads(); }
  const float sd  = sqrtf(red[0] / (float)(D - 1));
  const float inv = 1.0f / (sd + 1e-6f);
  for (int c = tid; c < D; c += 256) {
    float y = g[c] * (x[base + c] - mean) * inv + b[c];
    x[base + c]  = y;
    xB[base + c] = f32_to_bf16(y);
  }
}

__global__ __launch_bounds__(256)
void k_final_softmax(const float* __restrict__ x, float* __restrict__ out, int D) {
  __shared__ float red[256];
  const long base = (long)blockIdx.x * D;
  const int tid = threadIdx.x;
  float mx = -3.4e38f;
  for (int c = tid; c < D; c += 256) mx = fmaxf(mx, x[base + c]);
  red[tid] = mx; __syncthreads();
  for (int o = 128; o > 0; o >>= 1) { if (tid < o) red[tid] = fmaxf(red[tid], red[tid + o]); __syncthreads(); }
  mx = red[0]; __syncthreads();
  float s = 0.f;
  for (int c = tid; c < D; c += 256) s += __expf(x[base + c] - mx);
  red[tid] = s; __syncthreads();
  for (int o = 128; o > 0; o >>= 1) { if (tid < o) red[tid] += red[tid + o]; __syncthreads(); }
  const float inv = 1.0f / red[0];
  for (int c = tid; c < D; c += 256) out[base + c] = __expf(x[base + c] - mx) * inv;
}

// ---------------------------------------------------------------------------
// Host orchestration
// ---------------------------------------------------------------------------
extern "C" void kernel_launch(void* const* d_in, const int* in_sizes, int n_in,
                              void* d_out, int out_size, void* d_ws, size_t ws_size,
                              hipStream_t stream) {
  (void)in_sizes; (void)n_in; (void)out_size;

  const float* xl      = (const float*)d_in[0];
  const float* dl      = (const float*)d_in[1];
  const float* pe      = (const float*)d_in[2];
  const float* enc_wq  = (const float*)d_in[3];
  const float* enc_wk  = (const float*)d_in[4];
  const float* enc_wv  = (const float*)d_in[5];
  const float* enc_wo  = (const float*)d_in[6];
  const float* enc_bo  = (const float*)d_in[7];
  const float* enc_lna = (const float*)d_in[8];
  const float* enc_lnb = (const float*)d_in[9];
  const float* enc_w1  = (const float*)d_in[10];
  const float* enc_b1  = (const float*)d_in[11];
  const float* enc_w2  = (const float*)d_in[12];
  const float* enc_b2  = (const float*)d_in[13];
  const float* dec_wq  = (const float*)d_in[14];
  const float* dec_wk  = (const float*)d_in[15];
  const float* dec_wv  = (const float*)d_in[16];
  const float* dec_wo  = (const float*)d_in[17];
  const float* dec_bo  = (const float*)d_in[18];
  const float* dec_cwq = (const float*)d_in[19];
  const float* dec_cwk = (const float*)d_in[20];
  const float* dec_cwv = (const float*)d_in[21];
  const float* dec_cwo = (const float*)d_in[22];
  const float* dec_cbo = (const float*)d_in[23];
  const float* dec_lna = (const float*)d_in[24];
  const float* dec_lnb = (const float*)d_in[25];
  const float* dec_w1  = (const float*)d_in[26];
  const float* dec_b1  = (const float*)d_in[27];
  const float* dec_w2  = (const float*)d_in[28];
  const float* dec_b2  = (const float*)d_in[29];

  // ---- workspace carve-up ----
  char* p = (char*)d_ws;
  auto take = [&](size_t bytes) -> char* {
    char* r = p;
    p += (bytes + 255) & ~(size_t)255;
    return r;
  };
  float*          mF      = (float*)         take((size_t)SEQ * DMODEL * 4);
  unsigned short* mB      = (unsigned short*)take((size_t)SEQ * DMODEL * 2);
  float*          xF      = (float*)         take((size_t)SEQ * DMODEL * 4);
  unsigned short* xB      = (unsigned short*)take((size_t)SEQ * DMODEL * 2);
  unsigned short* qB      = (unsigned short*)take((size_t)SEQ * DMODEL * 2);
  unsigned short* kB      = (unsigned short*)take((size_t)SEQ * DMODEL * 2);
  unsigned short* vB      = (unsigned short*)take((size_t)SEQ * DMODEL * 2);
  unsigned short* vtB     = (unsigned short*)take((size_t)DMODEL * SEQ * 2);
  unsigned short* oB      = (unsigned short*)take((size_t)SEQ * DMODEL * 2);
  float*          fxF     = (float*)         take((size_t)SEQ * DMODEL * 4);
  unsigned short* hB      = (unsigned short*)take((size_t)SEQ * FFNH * 2);
  float*          scoresF = (float*)         take((size_t)NHEAD * SEQ * SEQ * 4);
  unsigned short* pB      = (unsigned short*)take((size_t)NHEAD * SEQ * SEQ * 2);
  unsigned short* wB      = (unsigned short*)take((size_t)FFNH * DMODEL * 2);
  if ((size_t)(p - (char*)d_ws) > ws_size) return;

  // ---- launch helpers ----
  auto conv = [&](const float* src, unsigned short* dst, long n) {
    k_f32_to_bf16<<<dim3((unsigned)((n + 255) / 256)), 256, 0, stream>>>(src, dst, (int)n);
  };
  auto gemm = [&](const unsigned short* Ax, const unsigned short* Bx, const float* bias,
                  float* C32, unsigned short* Cbf, int Mx, int Nx, int Kx,
                  int lda, int ldb, int ldc, long sA, long sB, long sC,
                  int batch, int relu) {
    dim3 g(Nx / BN, Mx / BM, batch);
    k_gemm_bf16nt<<<g, 256, 0, stream>>>(Ax, Bx, bias, C32, Cbf, Mx, Nx, Kx,
                                         lda, ldb, ldc, sA, sB, sC, relu);
  };
  auto mha = [&](const unsigned short* qin, const unsigned short* kvin,
                 const float* wq, const float* wk, const float* wv,
                 const float* wo, const float* bo, int causal) {
    conv(wq, wB, (long)DMODEL * DMODEL);
    gemm(qin, wB, nullptr, nullptr, qB, SEQ, DMODEL, DMODEL,
         DMODEL, DMODEL, DMODEL, 0, 0, 0, 1, 0);
    conv(wk, wB, (long)DMODEL * DMODEL);
    gemm(kvin, wB, nullptr, nullptr, kB, SEQ, DMODEL, DMODEL,
         DMODEL, DMODEL, DMODEL, 0, 0, 0, 1, 0);
    conv(wv, wB, (long)DMODEL * DMODEL);
    gemm(kvin, wB, nullptr, nullptr, vB, SEQ, DMODEL, DMODEL,
         DMODEL, DMODEL, DMODEL, 0, 0, 0, 1, 0);
    k_transpose_tv<<<dim3(SEQ * DMODEL / 256), 256, 0, stream>>>(vB, vtB);
    // scores[h] = Q_h K_h^T  (batched over heads; head = 64-col slice)
    gemm(qB, kB, nullptr, scoresF, nullptr, SEQ, SEQ, DHEAD,
         DMODEL, DMODEL, SEQ, DHEAD, DHEAD, (long)SEQ * SEQ, NHEAD, 0);
    k_attn_softmax<<<dim3(SEQ, NHEAD), 256, 0, stream>>>(scoresF, pB, SEQ, causal, 0.125f);
    // O_h = P_h V_h  (Vt is [head][e][t] so the NT GEMM is contiguous)
    gemm(pB, vtB, nullptr, nullptr, oB, SEQ, DHEAD, SEQ,
         SEQ, SEQ, DMODEL, (long)SEQ * SEQ, (long)DHEAD * SEQ, DHEAD, NHEAD, 0);
    conv(wo, wB, (long)DMODEL * DMODEL);
    gemm(oB, wB, bo, fxF, nullptr, SEQ, DMODEL, DMODEL,
         DMODEL, DMODEL, DMODEL, 0, 0, 0, 1, 0);
  };
  auto ffn = [&](const unsigned short* xin, const float* w1, const float* b1,
                 const float* w2, const float* b2) {
    conv(w1, wB, (long)FFNH * DMODEL);
    gemm(xin, wB, b1, nullptr, hB, SEQ, FFNH, DMODEL,
         DMODEL, DMODEL, FFNH, 0, 0, 0, 1, 1);
    conv(w2, wB, (long)DMODEL * FFNH);
    gemm(hB, wB, b2, fxF, nullptr, SEQ, DMODEL, FFNH,
         FFNH, FFNH, DMODEL, 0, 0, 0, 1, 0);
  };
  auto lnres = [&](float* xf, unsigned short* xb, const float* a, const float* b) {
    k_ln_res<<<dim3(SEQ), 256, 0, stream>>>(xf, xb, fxF, a, b, DMODEL);
  };

  const long WQK = (long)DMODEL * DMODEL;   // per-layer qkv/wo weight stride
  const long WF1 = (long)FFNH * DMODEL;

  // ---- encoder ----
  k_add_pos<<<dim3(SEQ), 256, 0, stream>>>(xl, pe, mF, mB, DMODEL);
  for (int l = 0; l < NLAYER; ++l) {
    mha(mB, mB, enc_wq + l * WQK, enc_wk + l * WQK, enc_wv + l * WQK,
        enc_wo + l * WQK, enc_bo + l * DMODEL, 0);
    lnres(mF, mB, enc_lna + (l * 2 + 0) * DMODEL, enc_lnb + (l * 2 + 0) * DMODEL);
    ffn(mB, enc_w1 + l * WF1, enc_b1 + l * FFNH, enc_w2 + l * WF1, enc_b2 + l * DMODEL);
    lnres(mF, mB, enc_lna + (l * 2 + 1) * DMODEL, enc_lnb + (l * 2 + 1) * DMODEL);
  }

  // ---- decoder ----
  k_add_pos<<<dim3(SEQ), 256, 0, stream>>>(dl, pe, xF, xB, DMODEL);
  for (int l = 0; l < NLAYER; ++l) {
    mha(xB, xB, dec_wq + l * WQK, dec_wk + l * WQK, dec_wv + l * WQK,
        dec_wo + l * WQK, dec_bo + l * DMODEL, 1);
    lnres(xF, xB, dec_lna + (l * 3 + 0) * DMODEL, dec_lnb + (l * 3 + 0) * DMODEL);
    mha(xB, mB, dec_cwq + l * WQK, dec_cwk + l * WQK, dec_cwv + l * WQK,
        dec_cwo + l * WQK, dec_cbo + l * DMODEL, 0);
    lnres(xF, xB, dec_lna + (l * 3 + 1) * DMODEL, dec_lnb + (l * 3 + 1) * DMODEL);
    ffn(xB, dec_w1 + l * WF1, dec_b1 + l * FFNH, dec_w2 + l * WF1, dec_b2 + l * DMODEL);
    lnres(xF, xB, dec_lna + (l * 3 + 2) * DMODEL, dec_lnb + (l * 3 + 2) * DMODEL);
  }

  k_final_softmax<<<dim3(SEQ), 256, 0, stream>>>(xF, (float*)d_out, DMODEL);
}
